// SSIMLoss_51049981280304
// MI455X (gfx1250) — matrix operand order repaired and checked by reading, not compiled, not run on hardware
//
#include <hip/hip_runtime.h>

typedef __attribute__((ext_vector_type(16))) _Float16 v16h;
typedef __attribute__((ext_vector_type(8)))  _Float16 v8h;
typedef __attribute__((ext_vector_type(8)))  float    v8f;

#define IMG_W 512
#define IMG_H 512
#define PLANES 96               // 32 batch * 3 channels
#define TILES_PER_PLANE 1024    // (512/16)^2
#define WAVES_PER_BLOCK 8
#define TOTAL_TILES (PLANES * TILES_PER_PLANE)          // 98304
#define GRID1 (TOTAL_TILES / WAVES_PER_BLOCK)           // 12288

// 11-tap Gaussian (sigma=1.5), normalized; padded with zeros so that
// index (k - mn + 16) is always in [0,48) for k in [0,32), mn in [0,16).
__device__ __constant__ float GWPAD[48] = {
    0.f,0.f,0.f,0.f, 0.f,0.f,0.f,0.f, 0.f,0.f,0.f,0.f, 0.f,0.f,0.f,0.f,
    0.00102841f, 0.00759872f, 0.03600077f, 0.10936110f, 0.21300553f,
    0.26601172f,
    0.21300553f, 0.10936110f, 0.03600077f, 0.00759872f, 0.00102841f,
    0.f,0.f,0.f,0.f,0.f, 0.f,0.f,0.f,0.f,0.f, 0.f,0.f,0.f,0.f,0.f,
    0.f,0.f,0.f,0.f,0.f, 0.f
};

static __device__ __forceinline__ v8f wmma16x16x32(v16h a, v16h b, v8f c) {
    return __builtin_amdgcn_wmma_f32_16x16x32_f16(
        /*neg_a=*/false, a, /*neg_b=*/false, b,
        /*c_mod=*/(short)0, c, /*reuse_a=*/false, /*reuse_b=*/false);
}

static __device__ __forceinline__ void quant5(float x1, float x2, int i,
                                              v16h& b1, v16h& b2, v16h& b11,
                                              v16h& b22, v16h& b12) {
    b1[i]  = (_Float16)x1;
    b2[i]  = (_Float16)x2;
    b11[i] = (_Float16)(x1 * x1);
    b22[i] = (_Float16)(x2 * x2);
    b12[i] = (_Float16)(x1 * x2);
}

// Load one 32-row x 16-col column block of both images in WMMA B layout:
// lane supplies column (c0-5 + cb*16 + lm), K rows i + 16*lhalf.
template <bool INTERIOR>
static __device__ __forceinline__ void load_cols(
    const float* __restrict__ p1, const float* __restrict__ p2,
    int r0, int c0, int cb, int lhalf, int lm,
    v16h& b1, v16h& b2, v16h& b11, v16h& b22, v16h& b12)
{
    const int col   = c0 - 5 + cb * 16 + lm;
    const int rbase = r0 - 5 + lhalf * 16;
    if (INTERIOR) {
        // all accesses in-bounds: straight immediate-offset load clause
        const float* q1 = p1 + (size_t)rbase * IMG_W + col;
        const float* q2 = p2 + (size_t)rbase * IMG_W + col;
#pragma unroll
        for (int i = 0; i < 16; ++i) {
            const float x1 = (q1[i * IMG_W] + 1.f) * 0.5f;
            const float x2 = (q2[i * IMG_W] + 1.f) * 0.5f;
            quant5(x1, x2, i, b1, b2, b11, b22, b12);
        }
    } else {
        // branchless edge handling: clamp address, select value
        // (SAME zero padding applies to x=(img+1)*0.5, so pad value is 0)
        const int  colc  = min(max(col, 0), IMG_W - 1);
        const bool colok = (col >= 0) & (col < IMG_W);
#pragma unroll
        for (int i = 0; i < 16; ++i) {
            const int  row  = rbase + i;
            const int  rowc = min(max(row, 0), IMG_H - 1);
            const bool ok   = colok & (row >= 0) & (row < IMG_H);
            const size_t off = (size_t)rowc * IMG_W + colc;
            const float r1 = p1[off];
            const float r2 = p2[off];
            const float x1 = ok ? (r1 + 1.f) * 0.5f : 0.f;
            const float x2 = ok ? (r2 + 1.f) * 0.5f : 0.f;
            quant5(x1, x2, i, b1, b2, b11, b22, b12);
        }
    }
}

__global__ __launch_bounds__(256)
void ssim_tile_kernel(const float* __restrict__ img1,
                      const float* __restrict__ img2,
                      float* __restrict__ partials)
{
    // per-wave 16x32 f16 intermediate, column-major [col][row], 16B aligned
    __shared__ __align__(16) _Float16 lbuf[WAVES_PER_BLOCK][520];
    __shared__ float red[256];

    const int tid   = threadIdx.x;
    const int lane  = tid & 31;
    const int wave  = tid >> 5;
    const int lhalf = lane >> 4;     // 0: lanes 0-15, 1: lanes 16-31
    const int lm    = lane & 15;

    const int tile_id = blockIdx.x * WAVES_PER_BLOCK + wave;   // < 98304
    const int plane   = tile_id >> 10;
    const int t       = tile_id & 1023;
    const int r0      = (t >> 5) << 4;
    const int c0      = (t & 31) << 4;
    const float* p1 = img1 + (size_t)plane * (IMG_W * IMG_H);
    const float* p2 = img2 + (size_t)plane * (IMG_W * IMG_H);

    // wave-uniform interior test: rows/cols accessed are [r0-5, r0+26]
    const bool interior = (r0 >= 5) & (r0 <= IMG_H - 27) &
                          (c0 >= 5) & (c0 <= IMG_W - 27);

    // ---- constant band-matrix fragments (Gaussian taps) ------------------
    // Vertical-pass A fragment (M=out row=lm):  Wv[m,k] = g[k-m], 0<=k-m<=10
    // Horizontal-pass B fragment (N=out col=lm): Wh[k,n] = g[k-n], 0<=k-n<=10
    v16h aW, bW;
#pragma unroll
    for (int i = 0; i < 16; ++i) {
        const int kA = i + lhalf * 16;                   // contiguous K (matches B)
        const int kB = i + lhalf * 16;                   // contiguous K (matches aV)
        aW[i] = (_Float16)GWPAD[kA - lm + 16];
        bW[i] = (_Float16)GWPAD[kB - lm + 16];
    }

    const v8f zero = {0.f,0.f,0.f,0.f,0.f,0.f,0.f,0.f};

    // ---- pass 1: vertical conv of 5 quantities over 2 column blocks ------
    v8f acc[5][2];
#pragma unroll
    for (int cb = 0; cb < 2; ++cb) {
        v16h b1, b2, b11, b22, b12;
        if (interior) {
            load_cols<true >(p1, p2, r0, c0, cb, lhalf, lm, b1, b2, b11, b22, b12);
        } else {
            load_cols<false>(p1, p2, r0, c0, cb, lhalf, lm, b1, b2, b11, b22, b12);
        }
        acc[0][cb] = wmma16x16x32(aW, b1,  zero);
        acc[1][cb] = wmma16x16x32(aW, b2,  zero);
        acc[2][cb] = wmma16x16x32(aW, b11, zero);
        acc[3][cb] = wmma16x16x32(aW, b22, zero);
        acc[4][cb] = wmma16x16x32(aW, b12, zero);
    }

    // ---- pass 2: LDS transpose + horizontal conv per quantity ------------
    _Float16* buf = &lbuf[wave][0];
    v8f res[5];
#pragma unroll
    for (int q = 0; q < 5; ++q) {
        // store intermediate: C/D layout lane holds rows (d + 8*lhalf), col lm
#pragma unroll
        for (int cb = 0; cb < 2; ++cb) {
            v8h h;
#pragma unroll
            for (int d = 0; d < 8; ++d) h[d] = (_Float16)acc[q][cb][d];
            const int colIdx = cb * 16 + lm;
            *(v8h*)(buf + colIdx * 16 + lhalf * 8) = h;   // 16B packed store
        }
        asm volatile("" ::: "memory");   // keep DS program order (in-order pipe)
        // reload as A fragment: row M=lm, contiguous K = i + 16*lhalf (matches bW)
        v16h aV;
#pragma unroll
        for (int i = 0; i < 16; ++i) {
            const int k = i + lhalf * 16;
            aV[i] = buf[k * 16 + lm];
        }
        asm volatile("" ::: "memory");
        res[q] = wmma16x16x32(aV, bW, zero);
    }

    // ---- elementwise SSIM + reduction ------------------------------------
    // lane holds output pixel (r0 + d + 8*lhalf, c0 + lm) for d=0..7,
    // aligned across all five quantities.
    float lsum = 0.f;
#pragma unroll
    for (int d = 0; d < 8; ++d) {
        const float mu1 = res[0][d], mu2 = res[1][d];
        const float e11 = res[2][d], e22 = res[3][d], e12 = res[4][d];
        const float mu1s = mu1 * mu1, mu2s = mu2 * mu2, m12 = mu1 * mu2;
        const float s1 = e11 - mu1s, s2 = e22 - mu2s, s12 = e12 - m12;
        const float C1 = 1e-4f, C2 = 9e-4f;
        const float num = (2.f * m12 + C1) * (2.f * s12 + C2);
        const float den = (mu1s + mu2s + C1) * (s1 + s2 + C2);
        lsum += num / den;
    }

    red[tid] = lsum;
    __syncthreads();
#pragma unroll
    for (int s = 128; s > 0; s >>= 1) {
        if (tid < s) red[tid] += red[tid + s];
        __syncthreads();
    }
    if (tid == 0) partials[blockIdx.x] = red[0];
}

__global__ __launch_bounds__(256)
void ssim_reduce_kernel(const float* __restrict__ partials, int n,
                        float* __restrict__ out)
{
    __shared__ double dred[256];
    double s = 0.0;
    for (int i = threadIdx.x; i < n; i += 256) s += (double)partials[i];
    dred[threadIdx.x] = s;
    __syncthreads();
#pragma unroll
    for (int k = 128; k > 0; k >>= 1) {
        if (threadIdx.x < k) dred[threadIdx.x] += dred[threadIdx.x + k];
        __syncthreads();
    }
    if (threadIdx.x == 0) {
        const double mean = dred[0] / (double)((size_t)PLANES * IMG_W * IMG_H);
        out[0] = (float)(1.0 - mean);
    }
}

extern "C" void kernel_launch(void* const* d_in, const int* in_sizes, int n_in,
                              void* d_out, int out_size, void* d_ws, size_t ws_size,
                              hipStream_t stream) {
    (void)in_sizes; (void)n_in; (void)out_size; (void)ws_size;
    const float* img1 = (const float*)d_in[0];
    const float* img2 = (const float*)d_in[1];
    float* partials = (float*)d_ws;                 // GRID1 floats = 48 KB
    ssim_tile_kernel<<<GRID1, 256, 0, stream>>>(img1, img2, partials);
    ssim_reduce_kernel<<<1, 256, 0, stream>>>(partials, GRID1, (float*)d_out);
}
